// NodeClassificationGAT_84172769068046
// MI455X (gfx1250) — compile-verified, matrix operand order checked by reading, not run on hardware
//
#include <hip/hip_runtime.h>
#include <math.h>

typedef __attribute__((ext_vector_type(16))) _Float16 v16h;
typedef __attribute__((ext_vector_type(8)))  float    v8f;
typedef __attribute__((ext_vector_type(4)))  float    v4f;

#define LRELU_SLOPE 0.2f

// ---------------------------------------------------------------------------
// Float atomic max via ordered-int encoding (monotone for IEEE-754 floats).
// ---------------------------------------------------------------------------
__device__ __forceinline__ void atomicMaxFloat(float* addr, float val) {
    if (val >= 0.0f) atomicMax((int*)addr, __float_as_int(val));
    else             atomicMin((unsigned int*)addr, __float_as_uint(val));
}

// Edge (src,dst) including synthesized self loops for e >= E.
__device__ __forceinline__ void edge_sd(const long long* __restrict__ ei,
                                        long long E, long long e, int& s, int& d) {
    if (e < E) { s = (int)ei[e]; d = (int)ei[E + e]; }
    else       { s = d = (int)(e - E); }
}

// ---------------------------------------------------------------------------
// Pack B[K,N] (f32, row-major) into per-wave WMMA B fragments (f16).
// Fragment f = kb*(N/16) + tn holds the 16x16x32 B tile for k-block kb,
// col-tile tn, laid out so lane l reads its 16 halfs contiguously:
//   Bf[(f*32 + l)*16 + e], e<8 -> K = kb*32 + kh*8 + e, col = tn*16 + (l&15)
//                          e>=8 -> K = kb*32 + 16 + kh*8 + (e-8)
// One 32-thread block per fragment.
// ---------------------------------------------------------------------------
__global__ void gat_pack_b(const float* __restrict__ B, _Float16* __restrict__ Bf,
                           int N, int K) {
    const int f  = blockIdx.x;
    const int l  = threadIdx.x & 31;
    const int NT = N >> 4;
    const int kb = f / NT;
    const int tn = f % NT;
    const int r  = l & 15;
    const int kh = l >> 4;
    const int col = tn * 16 + r;
    const int kLo = kb * 32 + kh * 8;
    const int kHi = kLo + 16;
    v16h frag;
#pragma unroll
    for (int e2 = 0; e2 < 8; ++e2) {
        frag[e2]     = (_Float16)B[(size_t)(kLo + e2) * N + col];
        frag[e2 + 8] = (_Float16)B[(size_t)(kHi + e2) * N + col];
    }
    ((v16h*)Bf)[(size_t)f * 32 + l] = frag;
}

// ---------------------------------------------------------------------------
// C[M, 16*NT chunk] = A[M,K] * B[K,N]  (f32 A/C, f16 WMMA, f32 accumulate)
// One wave per 16 rows x (16*NT) cols. A fragment loaded once per k-step
// (4x b128), B fragments stream from the packed buffer (2x b128 each),
// NT v_wmma per k-step. K % 32 == 0 required.
// A layout (ISA 7.12.2, 16-bit A 16x32): lane l holds row (l&15), kh=(l>>4):
//   elems 0..7 -> K = kh*8+{0..7}; elems 8..15 -> K = 16+kh*8+{0..7}
// C/D: elem v -> (M = v + 8*kh, N = lane&15).
// ---------------------------------------------------------------------------
template <int NT>
__global__ void gat_gemm_wmma(const float* __restrict__ A,
                              const _Float16* __restrict__ Bf,
                              float* __restrict__ C,
                              int M, int N, int K) {
    const int lane  = threadIdx.x & 31;
    const int r     = lane & 15;
    const int kh    = lane >> 4;
    const int chunk = blockIdx.x;            // N / (16*NT) chunks
    const int tM    = blockIdx.y;            // ceil(M/16) tiles
    const int NTtot = N >> 4;                // total col tiles in packed buffer

    int row = tM * 16 + r;
    if (row >= M) row = M - 1;               // clamp loads; stores guarded
    const float* __restrict__ Arow = A + (size_t)row * K;
    const v16h* __restrict__ Bfrag = (const v16h*)Bf;

    v8f acc[NT];
#pragma unroll
    for (int nt = 0; nt < NT; ++nt) acc[nt] = (v8f){};

    const int KB = K >> 5;
    for (int kb = 0; kb < KB; ++kb) {
        const int kLo = kb * 32 + kh * 8;    // multiple of 8 -> 32B aligned
        const int kHi = kLo + 16;

        const v4f lo0 = *(const v4f*)(Arow + kLo);
        const v4f lo1 = *(const v4f*)(Arow + kLo + 4);
        const v4f hi0 = *(const v4f*)(Arow + kHi);
        const v4f hi1 = *(const v4f*)(Arow + kHi + 4);
        v16h a;
#pragma unroll
        for (int e2 = 0; e2 < 4; ++e2) {
            a[e2]      = (_Float16)lo0[e2];
            a[e2 + 4]  = (_Float16)lo1[e2];
            a[e2 + 8]  = (_Float16)hi0[e2];
            a[e2 + 12] = (_Float16)hi1[e2];
        }

#pragma unroll
        for (int nt = 0; nt < NT; ++nt) {
            const int f = kb * NTtot + (chunk * NT + nt);
            const v16h b = Bfrag[(size_t)f * 32 + lane];
            acc[nt] = __builtin_amdgcn_wmma_f32_16x16x32_f16(
                false, a, false, b, (short)0, acc[nt], false, false);
        }
    }

    // ---- store: element v -> (m = tM*16 + v + 8*kh, n) ----
    const int mbase = tM * 16 + 8 * kh;
    if (tM * 16 + 16 <= M) {                 // full tile: unguarded
#pragma unroll
        for (int nt = 0; nt < NT; ++nt) {
            float* __restrict__ Cp = C + (size_t)mbase * N + (size_t)(chunk * NT + nt) * 16 + r;
#pragma unroll
            for (int v = 0; v < 8; ++v)
                Cp[(size_t)v * N] = acc[nt][v];
        }
    } else {                                 // tail tile
#pragma unroll
        for (int nt = 0; nt < NT; ++nt) {
            float* __restrict__ Cp = C + (size_t)mbase * N + (size_t)(chunk * NT + nt) * 16 + r;
#pragma unroll
            for (int v = 0; v < 8; ++v)
                if (mbase + v < M) Cp[(size_t)v * N] = acc[nt][v];
        }
    }
}

static inline void launch_gemm(const float* A, const _Float16* Bf, float* C,
                               int M, int N, int K, hipStream_t stream) {
    const unsigned mtiles = (unsigned)((M + 15) / 16);
    if (N % 128 == 0) {
        dim3 g(N / 128, mtiles);
        gat_gemm_wmma<8><<<g, 32, 0, stream>>>(A, Bf, C, M, N, K);
    } else if (N % 64 == 0) {
        dim3 g(N / 64, mtiles);
        gat_gemm_wmma<4><<<g, 32, 0, stream>>>(A, Bf, C, M, N, K);
    } else {
        dim3 g(N / 16, mtiles);
        gat_gemm_wmma<1><<<g, 32, 0, stream>>>(A, Bf, C, M, N, K);
    }
}

// ---------------------------------------------------------------------------
// Per-node attention halves: asrc[i] = h[i]·a_src, adst[i] = h[i]·a_dst
// ---------------------------------------------------------------------------
__global__ void gat_alphas(const float* __restrict__ h,
                           const float* __restrict__ a_src,
                           const float* __restrict__ a_dst,
                           float* __restrict__ asrc,
                           float* __restrict__ adst,
                           int Nn, int C) {
    const int i = blockIdx.x * blockDim.x + threadIdx.x;
    if (i >= Nn) return;
    const v4f* __restrict__ hi = (const v4f*)(h + (size_t)i * C);
    const v4f* __restrict__ as = (const v4f*)a_src;
    const v4f* __restrict__ ad = (const v4f*)a_dst;
    float s = 0.f, d = 0.f;
    const int C4 = C >> 2;
    for (int c = 0; c < C4; ++c) {
        const v4f v = hi[c], va = as[c], vb = ad[c];
#pragma unroll
        for (int j = 0; j < 4; ++j) { s += v[j] * va[j]; d += v[j] * vb[j]; }
    }
    asrc[i] = s;
    adst[i] = d;
}

__global__ void gat_init_nodes(float* __restrict__ mx, float* __restrict__ den, int Nn) {
    const int i = blockIdx.x * blockDim.x + threadIdx.x;
    if (i >= Nn) return;
    mx[i]  = -__builtin_inff();
    den[i] = 0.0f;
}

// agg[i*C + c] = bias[c]  (pre-seed so segment-sum lands on sum+bias)
__global__ void gat_init_bias(float* __restrict__ agg, const float* __restrict__ bias,
                              long long total, int C) {
    const long long t = (long long)blockIdx.x * blockDim.x + threadIdx.x;
    if (t >= total) return;
    agg[t] = bias[(int)(t % C)];
}

__global__ void gat_edge_max(const long long* __restrict__ ei, long long E, long long Etot,
                             const float* __restrict__ asrc, const float* __restrict__ adst,
                             float* __restrict__ logit, float* __restrict__ mx) {
    const long long e = (long long)blockIdx.x * blockDim.x + threadIdx.x;
    if (e >= Etot) return;
    int s, d; edge_sd(ei, E, e, s, d);
    float x = asrc[s] + adst[d];
    x = (x > 0.f) ? x : LRELU_SLOPE * x;
    logit[e] = x;
    atomicMaxFloat(&mx[d], x);
}

__global__ void gat_edge_exp(const long long* __restrict__ ei, long long E, long long Etot,
                             float* __restrict__ logit, const float* __restrict__ mx,
                             float* __restrict__ den) {
    const long long e = (long long)blockIdx.x * blockDim.x + threadIdx.x;
    if (e >= Etot) return;
    int s, d; edge_sd(ei, E, e, s, d);
    const float w = __expf(logit[e] - mx[d]);
    logit[e] = w;
    atomicAdd(&den[d], w);
}

// One block per edge, one thread per channel.
__global__ void gat_aggregate(const long long* __restrict__ ei, long long E, long long Etot,
                              const float* __restrict__ h, const float* __restrict__ w,
                              const float* __restrict__ den, float* __restrict__ out, int C) {
    const long long e = blockIdx.x;
    if (e >= Etot) return;
    int s, d; edge_sd(ei, E, e, s, d);
    const float coef = w[e] / den[d];
    const float* __restrict__ hs = h + (size_t)s * C;
    float* __restrict__ od = out + (size_t)d * C;
    for (int c = threadIdx.x; c < C; c += blockDim.x)
        atomicAdd(&od[c], hs[c] * coef);
}

__global__ void gat_relu(float* __restrict__ x, long long total) {
    const long long t = (long long)blockIdx.x * blockDim.x + threadIdx.x;
    if (t >= total) return;
    const float v = x[t];
    x[t] = v > 0.f ? v : 0.f;
}

// ---------------------------------------------------------------------------
extern "C" void kernel_launch(void* const* d_in, const int* in_sizes, int n_in,
                              void* d_out, int out_size, void* d_ws, size_t ws_size,
                              hipStream_t stream) {
    const float*     x      = (const float*)d_in[0];
    const long long* ei     = (const long long*)d_in[1];
    const float*     W1     = (const float*)d_in[2];
    const float*     a_src1 = (const float*)d_in[3];
    const float*     a_dst1 = (const float*)d_in[4];
    const float*     b1     = (const float*)d_in[5];
    const float*     W2     = (const float*)d_in[6];
    const float*     a_src2 = (const float*)d_in[7];
    const float*     a_dst2 = (const float*)d_in[8];
    const float*     b2     = (const float*)d_in[9];

    const int C_hid = in_sizes[3];                       // 128
    const int C_in  = in_sizes[2] / C_hid;               // 256
    const int Nn    = in_sizes[0] / C_in;                // 50000
    const long long E    = (long long)in_sizes[1] / 2;   // 640000
    const long long Etot = E + Nn;                       // + self loops
    const int C_out = in_sizes[7];                       // 64

    // ---- carve workspace, 256B aligned blocks ----
    size_t off = 0;
    auto carveB = [&](size_t bytes) {
        void* p = (char*)d_ws + off;
        off = (off + bytes + 255) & ~(size_t)255;
        return p;
    };
    auto carve = [&](size_t nflt) { return (float*)carveB(nflt * sizeof(float)); };

    float* H1   = carve((size_t)Nn * C_hid);
    float* HAGG = carve((size_t)Nn * C_hid);
    float* H2   = carve((size_t)Nn * C_out);
    float* AS1  = carve(Nn);
    float* AD1  = carve(Nn);
    float* M1   = carve(Nn);
    float* D1   = carve(Nn);
    float* AS2  = carve(Nn);
    float* AD2  = carve(Nn);
    float* M2   = carve(Nn);
    float* D2   = carve(Nn);
    float* EB1  = carve((size_t)Etot);
    float* EB2  = carve((size_t)Etot);
    const int nfrag1 = (C_in  / 32) * (C_hid / 16);      // 64 fragments
    const int nfrag2 = (C_hid / 32) * (C_out / 16);      // 16 fragments
    _Float16* BF1 = (_Float16*)carveB((size_t)nfrag1 * 512 * sizeof(_Float16));
    _Float16* BF2 = (_Float16*)carveB((size_t)nfrag2 * 512 * sizeof(_Float16));
    (void)ws_size; (void)n_in; (void)out_size;

    float* out = (float*)d_out;

    const int TB = 256;
    const unsigned nodeG = (unsigned)((Nn + TB - 1) / TB);
    const unsigned edgeG = (unsigned)((Etot + TB - 1) / TB);

    // ---- pack weights into WMMA fragment layout (tiny, L2-resident) ----
    gat_pack_b<<<nfrag1, 32, 0, stream>>>(W1, BF1, C_hid, C_in);
    gat_pack_b<<<nfrag2, 32, 0, stream>>>(W2, BF2, C_out, C_hid);

    // ================= Layer 1 =================
    launch_gemm(x, BF1, H1, Nn, C_hid, C_in, stream);
    gat_alphas<<<nodeG, TB, 0, stream>>>(H1, a_src1, a_dst1, AS1, AD1, Nn, C_hid);
    gat_init_nodes<<<nodeG, TB, 0, stream>>>(M1, D1, Nn);
    {
        const long long tot = (long long)Nn * C_hid;
        gat_init_bias<<<(unsigned)((tot + TB - 1) / TB), TB, 0, stream>>>(HAGG, b1, tot, C_hid);
    }
    gat_edge_max<<<edgeG, TB, 0, stream>>>(ei, E, Etot, AS1, AD1, EB1, M1);
    gat_edge_exp<<<edgeG, TB, 0, stream>>>(ei, E, Etot, EB1, M1, D1);
    gat_aggregate<<<(unsigned)Etot, 128, 0, stream>>>(ei, E, Etot, H1, EB1, D1, HAGG, C_hid);
    {
        const long long tot = (long long)Nn * C_hid;
        gat_relu<<<(unsigned)((tot + TB - 1) / TB), TB, 0, stream>>>(HAGG, tot);
    }

    // ================= Layer 2 =================
    launch_gemm(HAGG, BF2, H2, Nn, C_out, C_hid, stream);
    gat_alphas<<<nodeG, TB, 0, stream>>>(H2, a_src2, a_dst2, AS2, AD2, Nn, C_out);
    gat_init_nodes<<<nodeG, TB, 0, stream>>>(M2, D2, Nn);
    {
        const long long tot = (long long)Nn * C_out;
        gat_init_bias<<<(unsigned)((tot + TB - 1) / TB), TB, 0, stream>>>(out, b2, tot, C_out);
    }
    gat_edge_max<<<edgeG, TB, 0, stream>>>(ei, E, Etot, AS2, AD2, EB2, M2);
    gat_edge_exp<<<edgeG, TB, 0, stream>>>(ei, E, Etot, EB2, M2, D2);
    gat_aggregate<<<(unsigned)Etot, 64, 0, stream>>>(ei, E, Etot, H2, EB2, D2, out, C_out);
}
